// HybridClassifier_2534030705271
// MI455X (gfx1250) — compile-verified, matrix operand order checked by reading, not run on hardware
//
#include <hip/hip_runtime.h>

typedef __attribute__((ext_vector_type(16))) _Float16 v16h;
typedef __attribute__((ext_vector_type(8)))  _Float16 h8v;
typedef __attribute__((ext_vector_type(8)))  float    v8f;

// Model constants
constexpr int B_   = 16;
constexpr int L_   = 2048;
constexpr int TOK_ = B_ * L_;     // 32768 tokens
constexpr int H_   = 48;
constexpr int DM_  = 96;          // d_model = 2H
constexpr int ED_  = 768;         // 8 * d_model
constexpr int NS_  = 32;          // state size N
constexpr int DTR_ = 6;           // dt rank
constexpr int G4H_ = 192;         // 4*H gate width

// Branch-free fast transcendentals (v_exp_f32 + v_rcp_f32, no libm range code)
__device__ __forceinline__ float fsig(float x) {
  return __builtin_amdgcn_rcpf(1.0f + __expf(-x));
}
__device__ __forceinline__ float ftanh(float x) {
  return 2.0f * fsig(2.0f * x) - 1.0f;
}

// ---------------------------------------------------------------------------
// Weight pack: f32 W[N,K] -> f16 fragment-swizzled B operand.
// Layout: dst[(((kb*tilesN + tn)*32 + lane)*16 + i], i.e. each lane's 16
// halves of the 16x16x32 B-fragment are contiguous (two b128 loads in gemm).
// ---------------------------------------------------------------------------
__global__ void pack_wsw_kernel(const float* __restrict__ W, _Float16* __restrict__ dst,
                                int N, int K, int tilesN, int kBlocks) {
  long gid = (long)blockIdx.x * blockDim.x + threadIdx.x;
  if (gid >= (long)kBlocks * tilesN * 512) return;
  const int i = (int)(gid & 15);
  const int lane = (int)((gid >> 4) & 31);
  const long tb = gid >> 9;                  // kb*tilesN + tn
  const int tn = (int)(tb % tilesN);
  const int kb = (int)(tb / tilesN);
  const int col = lane & 15, half = lane >> 4;
  const int k = kb * 32 + ((i < 8) ? (half * 8 + i) : (16 + half * 8 + (i - 8)));
  const int n = tn * 16 + col;
  dst[gid] = (k < K && n < N) ? (_Float16)W[(size_t)n * K + k] : (_Float16)0.0f;
}

// t-major activation pack: dst row (t*16 + b) <- src token (b*L + t), f32->f16,
// zero padded to colsPad. Makes a GEMM M-tile == one timestep's batch tile.
__global__ void pack_tmajor_kernel(const float* __restrict__ src, _Float16* __restrict__ dst,
                                   int cols, int colsPad, long total) {
  long gid = (long)blockIdx.x * blockDim.x + threadIdx.x;
  if (gid >= total) return;
  const int c = (int)(gid % colsPad);
  const long row = gid / colsPad;            // t*16 + b
  const int b = (int)(row & 15);
  const long t = row >> 4;
  dst[gid] = (c < cols) ? (_Float16)src[((size_t)b * L_ + t) * cols + c]
                        : (_Float16)0.0f;
}

// ---------------------------------------------------------------------------
// WMMA GEMM: C[M,N] = A[M,K](f16 row-major) * Bsw (fragment-swizzled f16).
// One wave computes NT adjacent 16x16 N-tiles sharing the A fragment.
// MODE 0: plain   1: +bias[n]   2: +residual[m,n]   3: +bias, store D-tiles
// swizzled (tile*256 + lane*8 + r) for the LSTM consumer.
// ---------------------------------------------------------------------------
template <int MODE, int NT>
__global__ __launch_bounds__(256) void gemm_wmma(
    const _Float16* __restrict__ Ap, const _Float16* __restrict__ Bsw,
    float* __restrict__ Cp, const float* __restrict__ aux,
    int M, int N, int K, int lda, int ldc) {
  const int wave = threadIdx.x >> 5;
  const int lane = threadIdx.x & 31;
  const int tilesN = N >> 4;
  const int superN = tilesN / NT;
  const int super = blockIdx.x * 8 + wave;
  if (super >= (M >> 4) * superN) return;     // wave-uniform (EXEC all-1 for WMMA)
  const int tm = super / superN;
  const int tn0 = (super - tm * superN) * NT;
  const int rc = lane & 15, half = lane >> 4;
  const _Float16* Arow = Ap + (size_t)(tm * 16 + rc) * lda + half * 8;
  v8f acc[NT] = {};
  for (int k0 = 0; k0 < K; k0 += 32) {
    __builtin_prefetch(Arow + k0 + 64, 0, 1);           // global_prefetch_b8
    const h8v alo = *(const h8v*)(Arow + k0);
    const h8v ahi = *(const h8v*)(Arow + k0 + 16);
    v16h a;
#pragma unroll
    for (int i = 0; i < 8; ++i) { a[i] = alo[i]; a[i + 8] = ahi[i]; }
    const _Float16* Bt = Bsw + (((size_t)(k0 >> 5) * tilesN + tn0) * 32 + lane) * 16;
#pragma unroll
    for (int j = 0; j < NT; ++j) {
      const h8v blo = *(const h8v*)(Bt + (size_t)j * 512);
      const h8v bhi = *(const h8v*)(Bt + (size_t)j * 512 + 8);
      v16h b;
#pragma unroll
      for (int i = 0; i < 8; ++i) { b[i] = blo[i]; b[i + 8] = bhi[i]; }
      acc[j] = __builtin_amdgcn_wmma_f32_16x16x32_f16(false, a, false, b, (short)0,
                                                      acc[j], false, false);
    }
  }
  const int rbase = half * 8;
#pragma unroll
  for (int j = 0; j < NT; ++j) {
    const int tn = tn0 + j;
    if (MODE == 3) {
      const float bval = aux[tn * 16 + rc];
      float* o = Cp + (size_t)(tm * tilesN + tn) * 256 + lane * 8;
#pragma unroll
      for (int r = 0; r < 8; ++r) o[r] = acc[j][r] + bval;
    } else {
#pragma unroll
      for (int r = 0; r < 8; ++r) {
        const size_t orow = (size_t)(tm * 16 + rbase + r);
        float v = acc[j][r];
        if (MODE == 1) v += aux[tn * 16 + rc];
        if (MODE == 2) v += aux[orow * ldc + tn * 16 + rc];
        Cp[orow * ldc + tn * 16 + rc] = v;
      }
    }
  }
}

// Fragment-swizzled B load (two b128s), used for resident Whh.T fragments.
__device__ __forceinline__ v16h load_bsw(const _Float16* __restrict__ Bsw, int tilesN,
                                         int kb, int tn, int lane) {
  const _Float16* p = Bsw + (((size_t)kb * tilesN + tn) * 32 + lane) * 16;
  const h8v lo = *(const h8v*)p;
  const h8v hi = *(const h8v*)(p + 8);
  v16h b;
#pragma unroll
  for (int i = 0; i < 8; ++i) { b[i] = lo[i]; b[i + 8] = hi[i]; }
  return b;
}

// ---------------------------------------------------------------------------
// LSTM recurrence: per step g = xg_t + h @ Whh.T + bhh via WMMA (M=16 batch,
// K=48->64, N=192 = 12 tiles / 12 waves). xg arrives pre-swizzled D-tile
// layout so the C-fragment is two contiguous b128 loads per lane.
// grid.x = 2 (fwd/bwd), block = 384 threads (12 waves).
// ---------------------------------------------------------------------------
__global__ __launch_bounds__(384) void lstm_recurrent_kernel(
    const float* __restrict__ xgF, const float* __restrict__ xgB,
    const _Float16* __restrict__ WTf, const _Float16* __restrict__ WTb,
    const float* __restrict__ bhhF, const float* __restrict__ bhhB,
    float* __restrict__ hcat) {
  const int dir = blockIdx.x;
  const float* xg = dir ? xgB : xgF;
  const _Float16* WT = dir ? WTb : WTf;
  const float* bhh = dir ? bhhB : bhhF;
  const int dirOff = dir * H_;

  __shared__ _Float16 hA[16][64];   // h state as f16 A-operand image (48 padded to 64)
  __shared__ float cS[16][H_];      // cell state
  __shared__ float gS[16][G4H_];    // pre-activation gates

  const int tid = threadIdx.x;
  for (int i = tid; i < 16 * 64; i += 384) ((_Float16*)hA)[i] = (_Float16)0.0f;
  for (int i = tid; i < 16 * H_; i += 384) ((float*)cS)[i] = 0.0f;
  __syncthreads();

  const int wave = tid >> 5, lane = tid & 31;
  const int col = lane & 15, half = lane >> 4, rbase = half * 8;
  const int n0 = wave * 16;
  // Whh.T fragments resident across all 2048 steps
  const v16h bf0 = load_bsw(WT, 12, 0, wave, lane);
  const v16h bf1 = load_bsw(WT, 12, 1, wave, lane);
  const float bias = bhh[n0 + col];

  for (int s = 0; s < L_; ++s) {
    const int t = dir ? (L_ - 1 - s) : s;
    const float* xp = xg + (size_t)(t * 12 + wave) * 256 + lane * 8;
    const float4 c0 = *(const float4*)(xp);
    const float4 c1 = *(const float4*)(xp + 4);
    v8f acc;
    acc[0] = c0.x; acc[1] = c0.y; acc[2] = c0.z; acc[3] = c0.w;
    acc[4] = c1.x; acc[5] = c1.y; acc[6] = c1.z; acc[7] = c1.w;
    v16h a0, a1;
#pragma unroll
    for (int i = 0; i < 8; ++i) {
      a0[i]     = hA[col][half * 8 + i];
      a0[i + 8] = hA[col][16 + half * 8 + i];
      a1[i]     = hA[col][32 + half * 8 + i];
      a1[i + 8] = hA[col][48 + half * 8 + i];
    }
    acc = __builtin_amdgcn_wmma_f32_16x16x32_f16(false, a0, false, bf0, (short)0, acc,
                                                 false, false);
    acc = __builtin_amdgcn_wmma_f32_16x16x32_f16(false, a1, false, bf1, (short)0, acc,
                                                 false, false);
#pragma unroll
    for (int r = 0; r < 8; ++r) gS[rbase + r][n0 + col] = acc[r] + bias;
    __syncthreads();
#pragma unroll
    for (int q = 0; q < 2; ++q) {
      const int idx = tid + q * 384;                 // 768 = 16 batches * 48 hidden
      const int b = idx / H_, j = idx - b * H_;
      const float ig = gS[b][j], fg = gS[b][H_ + j];
      const float gg = gS[b][2 * H_ + j], og = gS[b][3 * H_ + j];
      const float c = fsig(fg) * cS[b][j] + fsig(ig) * ftanh(gg);
      const float hv = fsig(og) * ftanh(c);
      cS[b][j] = c;
      hA[b][j] = (_Float16)hv;
      hcat[(size_t)b * L_ * DM_ + (size_t)t * DM_ + dirOff + j] = hv;
    }
    __syncthreads();
  }
}

// ---------------------------------------------------------------------------
// RMSNorm: one wave per token (3 cols/lane), shfl_xor reduction, f16 out
// ---------------------------------------------------------------------------
__global__ __launch_bounds__(256) void rmsnorm_kernel(const float* __restrict__ hin,
                                                      const float* __restrict__ nw,
                                                      _Float16* __restrict__ xn) {
  const int lane = threadIdx.x & 31;
  const size_t tok = (size_t)blockIdx.x * 8 + (threadIdx.x >> 5);
  const float* hp = hin + tok * DM_;
  const float x0 = hp[lane], x1 = hp[lane + 32], x2 = hp[lane + 64];
  float ss = x0 * x0 + x1 * x1 + x2 * x2;
#pragma unroll
  for (int m = 16; m >= 1; m >>= 1) ss += __shfl_xor(ss, m, 32);
  const float r = rsqrtf(ss * (1.0f / DM_) + 1e-5f);
  _Float16* xp = xn + tok * DM_;
  xp[lane]      = (_Float16)(x0 * r * nw[lane]);
  xp[lane + 32] = (_Float16)(x1 * r * nw[lane + 32]);
  xp[lane + 64] = (_Float16)(x2 * r * nw[lane + 64]);
}

// ---------------------------------------------------------------------------
// Causal depthwise conv (K=16) + SiLU over cols[0:768) of (TOK,1536) buffer
// ---------------------------------------------------------------------------
__global__ void conv_silu_kernel(const float* __restrict__ xm,
                                 const float* __restrict__ cw,
                                 const float* __restrict__ cb,
                                 float* __restrict__ xc, _Float16* __restrict__ xch) {
  const size_t gid = (size_t)blockIdx.x * blockDim.x + threadIdx.x;
  if (gid >= (size_t)TOK_ * ED_) return;
  const int e = (int)(gid % ED_);
  const size_t tok = gid / ED_;
  const int t = (int)(tok % L_);
  float s = cb[e];
#pragma unroll
  for (int k = 0; k < 16; ++k) {
    const int tt = t - 15 + k;
    if (tt >= 0) s += xm[(size_t)(tok + k - 15) * 1536 + e] * cw[e * 16 + k];
  }
  const float v = s * fsig(s);
  xc[tok * ED_ + e] = v;
  xch[tok * ED_ + e] = (_Float16)v;
}

// delta = softplus(dbc[:, :6] @ dt_proj_w.T + dt_proj_b)
__global__ void dtproj_kernel(const float* __restrict__ dbc, const float* __restrict__ Wdt,
                              const float* __restrict__ bdt, float* __restrict__ delta) {
  const size_t gid = (size_t)blockIdx.x * blockDim.x + threadIdx.x;
  if (gid >= (size_t)TOK_ * ED_) return;
  const int e = (int)(gid % ED_);
  const size_t tok = gid / ED_;
  float s = bdt[e];
#pragma unroll
  for (int r = 0; r < DTR_; ++r) s += dbc[tok * 80 + r] * Wdt[e * DTR_ + r];
  delta[tok * ED_ + e] = (s > 20.0f) ? s : __logf(1.0f + __expf(s));
}

// ---------------------------------------------------------------------------
// Selective scan: lane-pair per (batch,channel); each thread owns 16 of the
// 32 states (+A row) in registers; partial y combined by shfl_xor. B_t/C_t
// double-buffered in LDS -> one barrier per step. grid = 16 batches * 3.
// ---------------------------------------------------------------------------
__global__ __launch_bounds__(512) void sel_scan_kernel(
    const float* __restrict__ xc, const float* __restrict__ delta,
    const float* __restrict__ dbc, const float* __restrict__ Alog,
    const float* __restrict__ Dp, float* __restrict__ y, int ldy) {
  const int tid = threadIdx.x;
  const int b = blockIdx.x / 3;
  const int el = tid >> 1, p = tid & 1;
  const int e = (blockIdx.x % 3) * 256 + el;
  const int nb = p * 16;
  __shared__ float Bs[2][NS_], Cs[2][NS_];
  float a[16], h[16];
  const float dskip = Dp[e];
#pragma unroll
  for (int n = 0; n < 16; ++n) {
    a[n] = -__expf(Alog[(size_t)e * NS_ + nb + n]);
    h[n] = 0.0f;
  }
  if (tid < 64) {
    const float v = dbc[(size_t)b * L_ * 80 + 6 + tid];  // cols 6..37 = B, 38..69 = C
    if (tid < 32) Bs[0][tid] = v; else Cs[0][tid - 32] = v;
  }
  __syncthreads();
  for (int t = 0; t < L_; ++t) {
    const size_t tok = (size_t)b * L_ + t;
    const int cur = t & 1;
    if (t + 1 < L_ && tid < 64) {
      const float v = dbc[(tok + 1) * 80 + 6 + tid];
      if (tid < 32) Bs[cur ^ 1][tid] = v; else Cs[cur ^ 1][tid - 32] = v;
    }
    const float xv = xc[tok * ED_ + e];
    const float dv = delta[tok * ED_ + e];
    const float dx = dv * xv;
    float accy = 0.0f;
#pragma unroll
    for (int n = 0; n < 16; ++n) {
      h[n] = __expf(dv * a[n]) * h[n] + dx * Bs[cur][nb + n];
      accy += h[n] * Cs[cur][nb + n];
    }
    accy += __shfl_xor(accy, 1, 32);
    if (p == 0) y[tok * ldy + e] = accy + xv * dskip;
    __syncthreads();
  }
}

// g = y * silu(z), packed to f16 for the out_proj gemm
__global__ void gate_kernel(const float* __restrict__ yz, _Float16* __restrict__ g) {
  const size_t gid = (size_t)blockIdx.x * blockDim.x + threadIdx.x;
  if (gid >= (size_t)TOK_ * ED_) return;
  const int e = (int)(gid % ED_);
  const size_t tok = gid / ED_;
  const float yv = yz[tok * 1536 + e];
  const float zv = yz[tok * 1536 + 768 + e];
  g[tok * ED_ + e] = (_Float16)(yv * (zv * fsig(zv)));
}

// Final FC: out[b,c] = h[b, L-1, :] . fc_w[c, :] + fc_b[c]
__global__ void fc_kernel(const float* __restrict__ h, const float* __restrict__ W,
                          const float* __restrict__ bias, float* __restrict__ out) {
  const int tid = threadIdx.x;
  if (tid >= B_ * 4) return;
  const int b = tid >> 2, c = tid & 3;
  const float* hp = h + ((size_t)b * L_ + (L_ - 1)) * DM_;
  float s = bias[c];
  for (int d = 0; d < DM_; ++d) s += hp[d] * W[c * DM_ + d];
  out[tid] = s;
}

// ---------------------------------------------------------------------------
// Host orchestration
// ---------------------------------------------------------------------------
extern "C" void kernel_launch(void* const* d_in, const int* in_sizes, int n_in,
                              void* d_out, int out_size, void* d_ws, size_t ws_size,
                              hipStream_t stream) {
  (void)in_sizes; (void)n_in; (void)out_size; (void)ws_size;
  const float* x = (const float*)d_in[0];
  const float* Wih[2][2] = {{(const float*)d_in[1], (const float*)d_in[5]},
                            {(const float*)d_in[9], (const float*)d_in[13]}};
  const float* Whh[2][2] = {{(const float*)d_in[2], (const float*)d_in[6]},
                            {(const float*)d_in[10], (const float*)d_in[14]}};
  const float* bih[2][2] = {{(const float*)d_in[3], (const float*)d_in[7]},
                            {(const float*)d_in[11], (const float*)d_in[15]}};
  const float* bhh[2][2] = {{(const float*)d_in[4], (const float*)d_in[8]},
                            {(const float*)d_in[12], (const float*)d_in[16]}};
  const float* norm_w = (const float*)d_in[17];
  const float* in_proj = (const float*)d_in[18];
  const float* conv_w = (const float*)d_in[19];
  const float* conv_b = (const float*)d_in[20];
  const float* x_proj = (const float*)d_in[21];
  const float* dt_w = (const float*)d_in[22];
  const float* dt_b = (const float*)d_in[23];
  const float* A_log = (const float*)d_in[24];
  const float* D_par = (const float*)d_in[25];
  const float* out_proj = (const float*)d_in[26];
  const float* fc_w = (const float*)d_in[27];
  const float* fc_b = (const float*)d_in[28];
  float* out = (float*)d_out;

  char* base = (char*)d_ws;
  size_t off = 0;
  auto alloc = [&](size_t bytes) -> char* {
    char* p = base + off;
    off += (bytes + 255) & ~(size_t)255;
    return p;
  };

  // f16 weight packs (fragment-swizzled)
  _Float16* WihT[2][2];
  WihT[0][0] = (_Float16*)alloc((size_t)1 * 12 * 512 * 2);
  WihT[0][1] = (_Float16*)alloc((size_t)1 * 12 * 512 * 2);
  WihT[1][0] = (_Float16*)alloc((size_t)3 * 12 * 512 * 2);
  WihT[1][1] = (_Float16*)alloc((size_t)3 * 12 * 512 * 2);
  _Float16* WhhT[2][2];
  for (int l = 0; l < 2; ++l)
    for (int d = 0; d < 2; ++d) WhhT[l][d] = (_Float16*)alloc((size_t)2 * 12 * 512 * 2);
  _Float16* inT[2];  for (int l = 0; l < 2; ++l) inT[l] = (_Float16*)alloc((size_t)3 * 96 * 512 * 2);
  _Float16* xpT[2];  for (int l = 0; l < 2; ++l) xpT[l] = (_Float16*)alloc((size_t)24 * 5 * 512 * 2);
  _Float16* opT[2];  for (int l = 0; l < 2; ++l) opT[l] = (_Float16*)alloc((size_t)24 * 6 * 512 * 2);

  // activation staging + f32 buffers
  _Float16* xpack = (_Float16*)alloc((size_t)TOK_ * 32 * 2);   // t-major x
  _Float16* actp  = (_Float16*)alloc((size_t)TOK_ * 96 * 2);   // t-major l1 input / rmsnorm out
  _Float16* st768 = (_Float16*)alloc((size_t)TOK_ * 768 * 2);
  float* xgF   = (float*)alloc((size_t)TOK_ * 192 * 4);        // swizzled D-tiles
  float* xgB   = (float*)alloc((size_t)TOK_ * 192 * 4);
  float* hbuf0 = (float*)alloc((size_t)TOK_ * 96 * 4);
  float* hbuf1 = (float*)alloc((size_t)TOK_ * 96 * 4);
  float* bufA  = (float*)alloc((size_t)TOK_ * 1536 * 4);  // in_proj out; y aliases xm slots
  float* bufB  = (float*)alloc((size_t)TOK_ * 768 * 4);   // conv+silu f32
  float* bufC  = (float*)alloc((size_t)TOK_ * 80 * 4);    // x_proj out (70->80)
  float* bufD  = (float*)alloc((size_t)TOK_ * 768 * 4);   // delta

  auto packW = [&](const float* W, _Float16* dst, int N, int K, int tilesN, int kBlocks) {
    long n = (long)kBlocks * tilesN * 512;
    pack_wsw_kernel<<<(unsigned)((n + 255) / 256), 256, 0, stream>>>(W, dst, N, K, tilesN, kBlocks);
  };
  for (int d = 0; d < 2; ++d) {
    packW(Wih[0][d], WihT[0][d], 192, 6, 12, 1);
    packW(Wih[1][d], WihT[1][d], 192, 96, 12, 3);
    packW(Whh[0][d], WhhT[0][d], 192, 48, 12, 2);
    packW(Whh[1][d], WhhT[1][d], 192, 48, 12, 2);
  }
  for (int l = 0; l < 2; ++l) {
    packW(in_proj + (size_t)l * 1536 * 96, inT[l], 1536, 96, 96, 3);
    packW(x_proj + (size_t)l * 70 * 768, xpT[l], 70, 768, 5, 24);
    packW(out_proj + (size_t)l * 96 * 768, opT[l], 96, 768, 6, 24);
  }

  // ---- LSTM layer 0 ----
  {
    long n = (long)TOK_ * 32;
    pack_tmajor_kernel<<<(unsigned)((n + 255) / 256), 256, 0, stream>>>(x, xpack, 6, 32, n);
  }
  // xg gemms: M=32768 (t-major), N=192, K=32; 12288 supertiles / 8 per block
  gemm_wmma<3, 2><<<1536, 256, 0, stream>>>(xpack, WihT[0][0], xgF, bih[0][0], TOK_, 192, 32, 32, 192);
  gemm_wmma<3, 2><<<1536, 256, 0, stream>>>(xpack, WihT[0][1], xgB, bih[0][1], TOK_, 192, 32, 32, 192);
  lstm_recurrent_kernel<<<2, 384, 0, stream>>>(xgF, xgB, WhhT[0][0], WhhT[0][1],
                                               bhh[0][0], bhh[0][1], hbuf0);
  // ---- LSTM layer 1 ----
  {
    long n = (long)TOK_ * 96;
    pack_tmajor_kernel<<<(unsigned)((n + 255) / 256), 256, 0, stream>>>(hbuf0, actp, 96, 96, n);
  }
  gemm_wmma<3, 2><<<1536, 256, 0, stream>>>(actp, WihT[1][0], xgF, bih[1][0], TOK_, 192, 96, 96, 192);
  gemm_wmma<3, 2><<<1536, 256, 0, stream>>>(actp, WihT[1][1], xgB, bih[1][1], TOK_, 192, 96, 96, 192);
  lstm_recurrent_kernel<<<2, 384, 0, stream>>>(xgF, xgB, WhhT[1][0], WhhT[1][1],
                                               bhh[1][0], bhh[1][1], hbuf1);

  // ---- Mamba blocks ----
  float* hcur = hbuf1;
  float* hnext = hbuf0;
  const long nED = (long)TOK_ * ED_;
  const unsigned gED = (unsigned)((nED + 255) / 256);
  for (int l = 0; l < 2; ++l) {
    rmsnorm_kernel<<<TOK_ / 8, 256, 0, stream>>>(hcur, norm_w + l * DM_, actp);
    // in_proj: M=32768, N=1536, K=96 -> 98304 supertiles (NT=2)
    gemm_wmma<0, 2><<<12288, 256, 0, stream>>>(actp, inT[l], bufA, nullptr, TOK_, 1536, 96, 96, 1536);
    conv_silu_kernel<<<gED, 256, 0, stream>>>(bufA, conv_w + (size_t)l * ED_ * 16,
                                              conv_b + l * ED_, bufB, st768);
    // x_proj: N=80 (5 tiles, NT=1), K=768 -> 10240 tiles
    gemm_wmma<0, 1><<<1280, 256, 0, stream>>>(st768, xpT[l], bufC, nullptr, TOK_, 80, 768, 768, 80);
    dtproj_kernel<<<gED, 256, 0, stream>>>(bufC, dt_w + (size_t)l * ED_ * DTR_,
                                           dt_b + l * ED_, bufD);
    sel_scan_kernel<<<48, 512, 0, stream>>>(bufB, bufD, bufC,
                                            A_log + (size_t)l * ED_ * NS_,
                                            D_par + l * ED_, bufA, 1536);
    gate_kernel<<<gED, 256, 0, stream>>>(bufA, st768);
    // out_proj + residual: N=96 (6 tiles, NT=2), K=768 -> 6144 supertiles
    gemm_wmma<2, 2><<<768, 256, 0, stream>>>(st768, opT[l], hnext, hcur, TOK_, 96, 768, 768, 96);
    float* tmp = hcur; hcur = hnext; hnext = tmp;
  }

  fc_kernel<<<1, 64, 0, stream>>>(hcur, fc_w, fc_b, out);
}
